// HybridGDN_32031866094021
// MI455X (gfx1250) — matrix operand order repaired and checked by reading, not compiled, not run on hardware
//
#include <hip/hip_runtime.h>
#include <hip/hip_bf16.h>

typedef __bf16 v16bf __attribute__((ext_vector_type(16)));
typedef float  v8f   __attribute__((ext_vector_type(8)));

// Set to 0 if the gfx1250 async-to-LDS asm fails to assemble.
#define USE_ASYNC_LDS 1

// ---------------- helpers ----------------
__device__ __forceinline__ __bf16 f2bf(float f) {
    unsigned int u = __builtin_bit_cast(unsigned int, f);
    unsigned int r = (u + 0x7FFFu + ((u >> 16) & 1u)) >> 16;
    return __builtin_bit_cast(__bf16, (unsigned short)r);
}
__device__ __forceinline__ float siluf(float x) { return x / (1.f + expf(-x)); }
__device__ __forceinline__ float sigmoidf_(float x) { return 1.f / (1.f + expf(-x)); }

// ---------------- embedding + hash n-grams ----------------
__launch_bounds__(256)
__global__ void k_embed(const float* __restrict__ tok_emb, const float* __restrict__ bigram,
                        const float* __restrict__ tri, const int* __restrict__ ids,
                        float* __restrict__ xcat, int T) {
    constexpr int D = 768, DB = 64, VB = 32000;
    int m = blockIdx.x;
    int t = m % T;
    int id    = ids[m];
    int prev  = (t > 0) ? ids[m - 1] : 0;
    int prev2 = (t > 1) ? ids[m - 2] : 0;
    int h1 = (prev * 131 + id) % VB;
    int h2 = (prev2 * 173 + prev * 131 + id) % VB;
    for (int c = threadIdx.x; c < D + DB; c += 256) {
        float v;
        if (c < D) v = tok_emb[(size_t)id * D + c];
        else {
            int d = c - D;
            v = bigram[(size_t)h1 * DB + d] + tri[(size_t)h2 * DB + d];
        }
        xcat[(size_t)m * (D + DB) + c] = v;
    }
}

// ---------------- rmsnorm (row-wise, optional weight) ----------------
__launch_bounds__(256)
__global__ void k_rmsnorm(const float* __restrict__ x, const float* __restrict__ w,
                          float* __restrict__ y, int D) {
    int m = blockIdx.x, tid = threadIdx.x;
    __shared__ float red[256];
    float s = 0.f;
    for (int c = tid; c < D; c += 256) { float v = x[(size_t)m * D + c]; s += v * v; }
    red[tid] = s; __syncthreads();
    for (int st = 128; st > 0; st >>= 1) { if (tid < st) red[tid] += red[tid + st]; __syncthreads(); }
    float inv = rsqrtf(red[0] / (float)D + 1e-6f);
    for (int c = tid; c < D; c += 256) {
        float v = x[(size_t)m * D + c] * inv;
        y[(size_t)m * D + c] = w ? v * w[c] : v;
    }
}

// ---------------- generic WMMA GEMM: C[M,N] (+)= alpha * A[M,K] @ W[K,N] ----------------
// A tile: staged as raw fp32 via gfx1250 GLOBAL_LOAD_ASYNC_TO_LDS_B128 (ASYNCcnt),
//         converted to bf16 at fragment build.
// W tile: float4 global loads, bf16-converted, stored transposed [n][k] for
//         contiguous B-fragment reads.
__launch_bounds__(256)
__global__ void k_gemm(const float* __restrict__ A, const float* __restrict__ W,
                       float* __restrict__ C, int M, int N, int K,
                       int addC, const float* __restrict__ alpha_ptr, int alpha_idx) {
    constexpr int BM = 64, BN = 128, KT = 32, KTP = 36, LDP = 40;
    __shared__ float  Asf[BM][KTP];   // fp32 A tile (async DMA target), padded
    __shared__ __bf16 Bs[BN][LDP];    // bf16 W tile, transposed
    const int tid  = threadIdx.x;
    const int lane = tid & 31;
    const int wave = tid >> 5;
    const int lo = lane & 15, hi = lane >> 4;
    const int m0 = blockIdx.y * BM;
    const int n0 = blockIdx.x * BN;
    const int wm = (wave >> 2) * 32;
    const int wn = (wave & 3) * 32;

    v8f acc[2][2];
    #pragma unroll
    for (int i = 0; i < 2; ++i)
        #pragma unroll
        for (int j = 0; j < 2; ++j)
            #pragma unroll
            for (int e = 0; e < 8; ++e) acc[i][j][e] = 0.f;

    for (int k0 = 0; k0 < K; k0 += KT) {
        // ---- stage A tile (64x32 fp32, 16B per lane per issue) ----
        #pragma unroll
        for (int i = 0; i < (BM * KT) / (256 * 4); ++i) {   // 2 issues/thread
            int idx4 = tid + 256 * i;
            int r  = idx4 >> 3;
            int c4 = (idx4 & 7) << 2;
            const float* gp = &A[(size_t)(m0 + r) * K + k0 + c4];
#if USE_ASYNC_LDS
            unsigned lds = (unsigned)(size_t)(void*)&Asf[r][c4];   // low 32b of flat LDS addr == LDS offset
            unsigned long long ga = (unsigned long long)(size_t)gp;
            asm volatile("global_load_async_to_lds_b128 %0, %1, off"
                         :: "v"(lds), "v"(ga) : "memory");
#else
            float4 a4 = *(const float4*)gp;
            Asf[r][c4 + 0] = a4.x; Asf[r][c4 + 1] = a4.y;
            Asf[r][c4 + 2] = a4.z; Asf[r][c4 + 3] = a4.w;
#endif
        }
        // ---- stage W tile (32x128 fp32 -> bf16 transposed) ----
        #pragma unroll
        for (int i = 0; i < (BN * KT) / (256 * 4); ++i) {   // 4 issues/thread
            int idx4 = tid + 256 * i;
            int n4 = (idx4 & 31) << 2;
            int kk = idx4 >> 5;
            float4 w4 = *(const float4*)&W[(size_t)(k0 + kk) * N + n0 + n4];
            Bs[n4 + 0][kk] = f2bf(w4.x);
            Bs[n4 + 1][kk] = f2bf(w4.y);
            Bs[n4 + 2][kk] = f2bf(w4.z);
            Bs[n4 + 3][kk] = f2bf(w4.w);
        }
        if (k0 + KT < K)   // gfx1250 global_prefetch_b8 of next W tile
            __builtin_prefetch(&W[(size_t)(k0 + KT) * N + n0 + (tid & (BN - 1))], 0, 1);
#if USE_ASYNC_LDS
        asm volatile("s_wait_asynccnt 0x0" ::: "memory");
#endif
        __syncthreads();

        v16bf a[2], b[2];
        #pragma unroll
        for (int i = 0; i < 2; ++i) {
            int row = wm + i * 16 + lo;
            #pragma unroll
            for (int e = 0; e < 16; ++e) {
                int kk = (hi << 4) + e;
                a[i][e] = f2bf(Asf[row][kk]);
            }
        }
        #pragma unroll
        for (int j = 0; j < 2; ++j) {
            int col = wn + j * 16 + lo;
            #pragma unroll
            for (int e = 0; e < 16; ++e) b[j][e] = Bs[col][(hi << 4) + e];
        }
        #pragma unroll
        for (int i = 0; i < 2; ++i)
            #pragma unroll
            for (int j = 0; j < 2; ++j)
                acc[i][j] = __builtin_amdgcn_wmma_f32_16x16x32_bf16(
                    false, a[i], false, b[j], (short)0, acc[i][j], false, false);
        __syncthreads();
    }

    float alpha = alpha_ptr ? alpha_ptr[alpha_idx] : 1.0f;
    #pragma unroll
    for (int i = 0; i < 2; ++i)
        #pragma unroll
        for (int j = 0; j < 2; ++j)
            #pragma unroll
            for (int r = 0; r < 8; ++r) {
                int row = m0 + wm + i * 16 + (hi << 3) + r;
                int col = n0 + wn + j * 16 + lo;
                size_t off = (size_t)row * N + col;
                float v = alpha * acc[i][j][r];
                C[off] = addC ? (C[off] + v) : v;
            }
}

// ---------------- beta/g small projections (N=12 each) ----------------
__launch_bounds__(256)
__global__ void k_beta_g(const float* __restrict__ xn, const float* __restrict__ Wb,
                         const float* __restrict__ Wa, const float* __restrict__ A_log,
                         const float* __restrict__ dt_bias,
                         float* __restrict__ beta, float* __restrict__ g) {
    constexpr int D = 768, H = 12;
    int m = blockIdx.x, tid = threadIdx.x;
    int wave = tid >> 5, lane = tid & 31;
    __shared__ float xs[D];
    for (int c = tid; c < D; c += 256) xs[c] = xn[(size_t)m * D + c];
    __syncthreads();
    for (int jj = 0; jj < 3; ++jj) {
        int j = wave * 3 + jj;                 // 0..23
        const float* W = (j < H) ? Wb : Wa;
        int col = (j < H) ? j : j - H;
        float p = 0.f;
        for (int k = lane; k < D; k += 32) p += xs[k] * W[(size_t)k * H + col];
        for (int off = 16; off > 0; off >>= 1) p += __shfl_down(p, off, 32);
        if (lane == 0) {
            if (j < H) beta[(size_t)m * H + col] = sigmoidf_(p);
            else {
                float z = p + dt_bias[col];
                float sp = (z > 20.f) ? z : log1pf(expf(z));
                g[(size_t)m * H + col] = -expf(A_log[col]) * sp;
            }
        }
    }
}

// ---------------- short conv (4-tap causal) + silu + optional l2norm ----------------
// mode: 0 = v (silu only), 1 = k (silu + l2norm), 2 = q (silu + l2norm + DK^-0.5)
__launch_bounds__(64)
__global__ void k_conv_silu(const float* __restrict__ in, const float* __restrict__ w,
                            float* __restrict__ out, int T, int mode) {
    constexpr int H = 12, HD = 768, KC = 4;
    __shared__ float redv[64];
    int bid = blockIdx.x;
    int h = bid % H; int t = (bid / H) % T; int b = bid / (H * T);
    int d = threadIdx.x;
    int c = h * 64 + d;
    float acc = 0.f;
    #pragma unroll
    for (int i = 0; i < KC; ++i) {
        int ts = t + i - (KC - 1);
        if (ts >= 0) acc += in[((size_t)b * T + ts) * HD + c] * w[c * KC + i];
    }
    float s = siluf(acc);
    float outv = s;
    if (mode > 0) {
        redv[d] = s * s;
        __syncthreads();
        for (int st = 32; st > 0; st >>= 1) { if (d < st) redv[d] += redv[d + st]; __syncthreads(); }
        float scale = rsqrtf(redv[0] + 1e-6f);
        if (mode == 2) scale *= 0.125f;        // DK^-0.5, DK=64
        outv = s * scale;
    }
    out[((size_t)b * T + t) * HD + c] = outv;
}

// ---------------- gated delta-rule scan (sequential over T, per (b,h)) ----------------
__launch_bounds__(256)
__global__ void k_delta_scan(const float* __restrict__ qc, const float* __restrict__ kc,
                             const float* __restrict__ vc, const float* __restrict__ beta,
                             const float* __restrict__ g, float* __restrict__ o, int T) {
    constexpr int H = 12, HD = 768;
    int bh = blockIdx.x;
    int b = bh / H, h = bh % H;
    int tid = threadIdx.x;
    int vcol = tid & 63;
    int kg = tid >> 6;                    // 0..3 (each owns 16 k rows)
    __shared__ float sk[64], sq[64], sv[64];
    __shared__ float red[4][64];
    __shared__ float sb, sg;
    float S[16];
    #pragma unroll
    for (int i = 0; i < 16; ++i) S[i] = 0.f;
    size_t base = (size_t)b * T * HD + h * 64;
    size_t bgb  = (size_t)b * T * H + h;
    for (int t = 0; t < T; ++t) {
        size_t off = base + (size_t)t * HD;
        if (tid < 64) {
            sk[tid] = kc[off + tid];
            sq[tid] = qc[off + tid];
            sv[tid] = vc[off + tid];
        }
        if (tid == 0) {
            sb = beta[bgb + (size_t)t * H];
            sg = expf(g[bgb + (size_t)t * H]);
        }
        __syncthreads();
        float eg = sg, bt = sb;
        float kvp = 0.f;
        #pragma unroll
        for (int i = 0; i < 16; ++i) { S[i] *= eg; kvp += sk[kg * 16 + i] * S[i]; }
        red[kg][vcol] = kvp;
        __syncthreads();
        float kv = red[0][vcol] + red[1][vcol] + red[2][vcol] + red[3][vcol];
        float delta = bt * (sv[vcol] - kv);
        float op = 0.f;
        #pragma unroll
        for (int i = 0; i < 16; ++i) { S[i] += sk[kg * 16 + i] * delta; op += sq[kg * 16 + i] * S[i]; }
        __syncthreads();
        red[kg][vcol] = op;
        __syncthreads();
        if (kg == 0)
            o[off + vcol] = red[0][vcol] + red[1][vcol] + red[2][vcol] + red[3][vcol];
        __syncthreads();
    }
}

// ---------------- per-head rmsnorm(o) * silu(gate) ----------------
__launch_bounds__(64)
__global__ void k_gate_norm(const float* __restrict__ o, const float* __restrict__ gate,
                            const float* __restrict__ w, float* __restrict__ out) {
    constexpr int H = 12, HD = 768;
    __shared__ float redv[64];
    int bid = blockIdx.x;
    int h = bid % H; int m = bid / H;
    int d = threadIdx.x;
    size_t off = (size_t)m * HD + h * 64 + d;
    float x = o[off];
    redv[d] = x * x;
    __syncthreads();
    for (int st = 32; st > 0; st >>= 1) { if (d < st) redv[d] += redv[d + st]; __syncthreads(); }
    float y = x * rsqrtf(redv[0] / 64.f + 1e-6f) * w[d];
    float gv = gate[off];
    out[off] = y * siluf(gv);
}

// ---------------- silu(u) * v for MLP ----------------
__launch_bounds__(256)
__global__ void k_silu_mul(const float* __restrict__ uv, float* __restrict__ h,
                           int Mrows, int HM) {
    size_t i = (size_t)blockIdx.x * 256 + threadIdx.x;
    size_t total = (size_t)Mrows * HM;
    if (i >= total) return;
    size_t m = i / HM, c = i % HM;
    float u = uv[m * (size_t)(2 * HM) + c];
    float v = uv[m * (size_t)(2 * HM) + HM + c];
    h[i] = siluf(u) * v;
}

// ---------------- fused logits (WMMA) + softcap + sum(exp) + target gather ----------------
__launch_bounds__(256)
__global__ void k_logits_loss(const float* __restrict__ Xn, const float* __restrict__ Emb,
                              const int* __restrict__ targets,
                              float* __restrict__ rowsum, float* __restrict__ tgt) {
    constexpr int RM = 32, CN = 128, KT = 32, D = 768, V = 32000, XLD = 776, ELD = 40;
    __shared__ __bf16 Xs[RM][XLD];
    __shared__ __bf16 Es[CN][ELD];
    __shared__ float rsum[RM];
    int tid = threadIdx.x; int lane = tid & 31; int wave = tid >> 5;
    int lo = lane & 15, hi = lane >> 4;
    int m0 = blockIdx.x * RM;
    int msub = (wave >> 2) * 16;
    int nsub = (wave & 3) * 32;

    for (int idx4 = tid; idx4 < RM * (D / 4); idx4 += 256) {
        int r  = idx4 / (D / 4);
        int c4 = (idx4 % (D / 4)) * 4;
        float4 v4 = *(const float4*)&Xn[(size_t)(m0 + r) * D + c4];
        Xs[r][c4 + 0] = f2bf(v4.x); Xs[r][c4 + 1] = f2bf(v4.y);
        Xs[r][c4 + 2] = f2bf(v4.z); Xs[r][c4 + 3] = f2bf(v4.w);
    }
    if (tid < RM) rsum[tid] = 0.f;
    __syncthreads();

    float rexp[8];
    int rowg[8]; int targ[8];
    #pragma unroll
    for (int r = 0; r < 8; ++r) {
        rexp[r] = 0.f;
        rowg[r] = m0 + msub + (hi << 3) + r;
        targ[r] = targets[rowg[r]];
    }

    for (int n0 = 0; n0 < V; n0 += CN) {
        v8f acc[2];
        #pragma unroll
        for (int j = 0; j < 2; ++j)
            #pragma unroll
            for (int e = 0; e < 8; ++e) acc[j][e] = 0.f;

        for (int k0 = 0; k0 < D; k0 += KT) {
            #pragma unroll
            for (int i = 0; i < (CN * KT) / (256 * 4); ++i) {   // 4
                int idx4 = tid + 256 * i;
                int n  = idx4 >> 3;
                int c4 = (idx4 & 7) << 2;
                float4 e4 = *(const float4*)&Emb[(size_t)(n0 + n) * D + k0 + c4];
                Es[n][c4 + 0] = f2bf(e4.x); Es[n][c4 + 1] = f2bf(e4.y);
                Es[n][c4 + 2] = f2bf(e4.z); Es[n][c4 + 3] = f2bf(e4.w);
            }
            __syncthreads();
            v16bf a; v16bf b[2];
            int arow = msub + lo;
            #pragma unroll
            for (int e = 0; e < 16; ++e) {
                int kk = (hi << 4) + e;
                a[e] = Xs[arow][k0 + kk];
            }
            #pragma unroll
            for (int j = 0; j < 2; ++j) {
                int col = nsub + j * 16 + lo;
                #pragma unroll
                for (int e = 0; e < 16; ++e) b[j][e] = Es[col][(hi << 4) + e];
            }
            #pragma unroll
            for (int j = 0; j < 2; ++j)
                acc[j] = __builtin_amdgcn_wmma_f32_16x16x32_bf16(
                    false, a, false, b[j], (short)0, acc[j], false, false);
            __syncthreads();
        }
        #pragma unroll
        for (int j = 0; j < 2; ++j) {
            int col = n0 + nsub + j * 16 + lo;
            #pragma unroll
            for (int r = 0; r < 8; ++r) {
                float lg = 30.0f * tanhf(acc[j][r] * (1.0f / 30.0f));
                rexp[r] += expf(lg);              // softcap bounds => no max-sub needed
                if (col == targ[r]) tgt[rowg[r]] = lg;
            }
        }
    }
    #pragma unroll
    for (int r = 0; r < 8; ++r) atomicAdd(&rsum[msub + (hi << 3) + r], rexp[r]);
    __syncthreads();
    if (tid < RM) rowsum[m0 + tid] = rsum[tid];   // block owns all V for its rows
}

// ---------------- final mean NLL ----------------
__launch_bounds__(256)
__global__ void k_finalize(const float* __restrict__ rowsum, const float* __restrict__ tgt,
                           float* __restrict__ out, int Mrows) {
    __shared__ float red[256];
    int tid = threadIdx.x;
    float s = 0.f;
    for (int i = tid; i < Mrows; i += 256) s += logf(rowsum[i]) - tgt[i];
    red[tid] = s; __syncthreads();
    for (int st = 128; st > 0; st >>= 1) { if (tid < st) red[tid] += red[tid + st]; __syncthreads(); }
    if (tid == 0) out[0] = red[0] / (float)Mrows;
}

// ---------------- host orchestration ----------------
extern "C" void kernel_launch(void* const* d_in, const int* in_sizes, int n_in,
                              void* d_out, int out_size, void* d_ws, size_t ws_size,
                              hipStream_t stream) {
    (void)in_sizes; (void)n_in; (void)out_size; (void)ws_size;
    const float* tok_emb      = (const float*)d_in[0];
    const float* bigram_w     = (const float*)d_in[1];
    const float* tri_w        = (const float*)d_in[2];
    const float* embed_proj_w = (const float*)d_in[3];
    const float* attn_norm_w  = (const float*)d_in[4];
    const float* mlp_norm_w   = (const float*)d_in[5];
    const float* Wq           = (const float*)d_in[6];
    const float* Wk           = (const float*)d_in[7];
    const float* Wv           = (const float*)d_in[8];
    const float* Wb           = (const float*)d_in[9];
    const float* Wa           = (const float*)d_in[10];
    const float* A_log        = (const float*)d_in[11];
    const float* dt_bias      = (const float*)d_in[12];
    const float* conv_q       = (const float*)d_in[13];
    const float* conv_k       = (const float*)d_in[14];
    const float* conv_v       = (const float*)d_in[15];
    const float* Wg           = (const float*)d_in[16];
    const float* o_norm_w     = (const float*)d_in[17];
    const float* Wo           = (const float*)d_in[18];
    const float* attn_scale   = (const float*)d_in[19];
    const float* mlp_scale    = (const float*)d_in[20];
    const float* mlp_fc_w     = (const float*)d_in[21];
    const float* mlp_proj_w   = (const float*)d_in[22];
    const float* final_norm_w = (const float*)d_in[23];
    const int*   input_ids    = (const int*)d_in[24];
    const int*   target_ids   = (const int*)d_in[25];

    constexpr int B = 2, T = 1024, Mtok = B * T, D = 768, DB = 64, H = 12, HM = 3072, L = 4;

    float* ws = (float*)d_ws;
    size_t p = 0;
    auto alloc = [&](size_t n) { float* r = ws + p; p += n; return r; };
    float* xcat   = alloc((size_t)Mtok * (D + DB));
    float* x      = alloc((size_t)Mtok * D);
    float* xn     = alloc((size_t)Mtok * D);
    float* qp     = alloc((size_t)Mtok * D);
    float* kp     = alloc((size_t)Mtok * D);
    float* vp     = alloc((size_t)Mtok * D);
    float* qc     = alloc((size_t)Mtok * D);
    float* kc     = alloc((size_t)Mtok * D);
    float* vc     = alloc((size_t)Mtok * D);
    float* gatebuf= alloc((size_t)Mtok * D);
    float* obuf   = alloc((size_t)Mtok * D);
    float* og     = alloc((size_t)Mtok * D);
    float* betab  = alloc((size_t)Mtok * H);
    float* gb     = alloc((size_t)Mtok * H);
    float* uv     = alloc((size_t)Mtok * 2 * HM);
    float* hbuf   = alloc((size_t)Mtok * HM);
    float* rowsum = alloc(Mtok);
    float* tgtbuf = alloc(Mtok);

    auto gemm = [&](const float* A, const float* W, float* C, int M_, int N_, int K_,
                    int addC, const float* ap, int ai) {
        dim3 g(N_ / 128, M_ / 64);
        k_gemm<<<g, 256, 0, stream>>>(A, W, C, M_, N_, K_, addC, ap, ai);
    };

    // embedding path
    k_embed<<<Mtok, 256, 0, stream>>>(tok_emb, bigram_w, tri_w, input_ids, xcat, T);
    gemm(xcat, embed_proj_w, x, Mtok, D, D + DB, 0, nullptr, 0);
    k_rmsnorm<<<Mtok, 256, 0, stream>>>(x, nullptr, x, D);   // in-place, no weight

    for (int l = 0; l < L; ++l) {
        k_rmsnorm<<<Mtok, 256, 0, stream>>>(x, attn_norm_w + (size_t)l * D, xn, D);
        gemm(xn, Wq + (size_t)l * D * D, qp, Mtok, D, D, 0, nullptr, 0);
        gemm(xn, Wk + (size_t)l * D * D, kp, Mtok, D, D, 0, nullptr, 0);
        gemm(xn, Wv + (size_t)l * D * D, vp, Mtok, D, D, 0, nullptr, 0);
        gemm(xn, Wg + (size_t)l * D * D, gatebuf, Mtok, D, D, 0, nullptr, 0);
        k_beta_g<<<Mtok, 256, 0, stream>>>(xn, Wb + (size_t)l * D * H, Wa + (size_t)l * D * H,
                                           A_log + l * H, dt_bias + l * H, betab, gb);
        int nconv = B * T * H;
        k_conv_silu<<<nconv, 64, 0, stream>>>(qp, conv_q + (size_t)l * D * 4, qc, T, 2);
        k_conv_silu<<<nconv, 64, 0, stream>>>(kp, conv_k + (size_t)l * D * 4, kc, T, 1);
        k_conv_silu<<<nconv, 64, 0, stream>>>(vp, conv_v + (size_t)l * D * 4, vc, T, 0);
        k_delta_scan<<<B * H, 256, 0, stream>>>(qc, kc, vc, betab, gb, obuf, T);
        k_gate_norm<<<Mtok * H, 64, 0, stream>>>(obuf, gatebuf, o_norm_w + (size_t)l * 64, og);
        gemm(og, Wo + (size_t)l * D * D, x, Mtok, D, D, 1, attn_scale, l);

        k_rmsnorm<<<Mtok, 256, 0, stream>>>(x, mlp_norm_w + (size_t)l * D, xn, D);
        gemm(xn, mlp_fc_w + (size_t)l * D * (2 * HM), uv, Mtok, 2 * HM, D, 0, nullptr, 0);
        size_t nsil = (size_t)Mtok * HM;
        k_silu_mul<<<(unsigned)((nsil + 255) / 256), 256, 0, stream>>>(uv, hbuf, Mtok, HM);
        gemm(hbuf, mlp_proj_w + (size_t)l * HM * D, x, Mtok, D, HM, 1, mlp_scale, l);
    }

    k_rmsnorm<<<Mtok, 256, 0, stream>>>(x, final_norm_w, xn, D);
    k_logits_loss<<<Mtok / 32, 256, 0, stream>>>(xn, tok_emb, target_ids, rowsum, tgtbuf);
    k_finalize<<<1, 256, 0, stream>>>(rowsum, tgtbuf, (float*)d_out, Mtok);
}
